// DGT_5987184411120
// MI455X (gfx1250) — matrix operand order repaired and checked by reading, not compiled, not run on hardware
//
#include <hip/hip_runtime.h>
#include <hip/hip_bf16.h>
#include <math.h>

typedef __bf16 bf16_t;
typedef __attribute__((ext_vector_type(16))) __bf16 v16bf;
typedef __attribute__((ext_vector_type(8)))  __bf16 v8bf;
typedef __attribute__((ext_vector_type(8)))  float  v8f;
typedef __attribute__((address_space(3))) const char lds_char;

static constexpr int kBS = 16;
static constexpr int kN  = 1024;   // nodes (= timestep emb dim)
static constexpr int kD  = 128;    // per-head dim / out dim
static constexpr int kH  = 8;
static constexpr int kDH = kD * kH;      // 1024
static constexpr int kM  = kBS * kN;     // 16384 flattened rows

// ---------- helpers ----------
__device__ inline bf16_t f2bf(float f) {
  unsigned u = __float_as_uint(f);
  unsigned r = (u + 0x7fffu + ((u >> 16) & 1u)) >> 16;   // RNE
  unsigned short s = (unsigned short)r;
  bf16_t b;
  __builtin_memcpy(&b, &s, 2);
  return b;
}

__device__ inline v8f v8f_zero() {
  v8f z;
#pragma unroll
  for (int i = 0; i < 8; ++i) z[i] = 0.f;
  return z;
}

__device__ inline v8f wmma_bf16(v16bf a, v16bf b, v8f c) {
  return __builtin_amdgcn_wmma_f32_16x16x32_bf16(false, a, false, b, (short)0, c,
                                                 false, false);
}

// A fragment: 16x32 bf16 (M x K). Lane L: row = L&15, K chunks
// [koff, koff+8) and [16+koff, 16+koff+8), koff = (L>>4)*8.  (ISA 7.12.2)
__device__ inline v16bf load_frag_a(const bf16_t* base, int ld, int lane) {
  const int row = lane & 15, koff = (lane >> 4) * 8;
  const bf16_t* p = base + row * ld;
  v8bf lo = *(const v8bf*)(p + koff);
  v8bf hi = *(const v8bf*)(p + 16 + koff);
  v16bf r;
#pragma unroll
  for (int i = 0; i < 8; ++i) { r[i] = lo[i]; r[8 + i] = hi[i]; }
  return r;
}

// B fragment: 32x16 bf16 (K x N) from row-major Bt (N x K).
// Lane L: column = L&15 (-> Bt row), K chunks [koff, koff+8) and [16+koff, 16+koff+8), koff = (L>>4)*8.
__device__ inline v16bf load_frag_b(const bf16_t* base, int ld, int lane) {
  const int col = lane & 15, koff = (lane >> 4) * 8;
  const bf16_t* p = base + col * ld + koff;
  v8bf lo = *(const v8bf*)(p);
  v8bf hi = *(const v8bf*)(p + 16);
  v16bf r;
#pragma unroll
  for (int i = 0; i < 8; ++i) { r[i] = lo[i]; r[8 + i] = hi[i]; }
  return r;
}

// Wave computes a 32x64 f32 tile (2 row-frags x 4 col-frags, 8 WMMAs/K-step).
__device__ inline void gemm_2x4(const bf16_t* A, int lda, const bf16_t* Bt,
                                int ldb, int ksteps, int lane, v8f acc[8]) {
  for (int s = 0; s < ksteps; ++s) {
    v16bf a0 = load_frag_a(A + s * 32, lda, lane);
    v16bf a1 = load_frag_a(A + 16 * lda + s * 32, lda, lane);
    v16bf b0 = load_frag_b(Bt + s * 32, ldb, lane);
    v16bf b1 = load_frag_b(Bt + 16 * ldb + s * 32, ldb, lane);
    v16bf b2 = load_frag_b(Bt + 32 * ldb + s * 32, ldb, lane);
    v16bf b3 = load_frag_b(Bt + 48 * ldb + s * 32, ldb, lane);
    acc[0] = wmma_bf16(a0, b0, acc[0]);
    acc[1] = wmma_bf16(a0, b1, acc[1]);
    acc[2] = wmma_bf16(a0, b2, acc[2]);
    acc[3] = wmma_bf16(a0, b3, acc[3]);
    acc[4] = wmma_bf16(a1, b0, acc[4]);
    acc[5] = wmma_bf16(a1, b1, acc[5]);
    acc[6] = wmma_bf16(a1, b2, acc[6]);
    acc[7] = wmma_bf16(a1, b3, acc[7]);
  }
}

// ---------- gfx1250 async copy to LDS ----------
__device__ inline unsigned lds_byte_addr(const void* p) {
  return (unsigned)(unsigned long long)(lds_char*)p;   // addrspacecast -> LDS offset
}
__device__ inline void async_copy16(unsigned lds_byte, const void* gsrc) {
  asm volatile("global_load_async_to_lds_b128 %0, %1, off"
               :: "v"(lds_byte), "v"(gsrc) : "memory");
}
__device__ inline void wait_async0() {
  asm volatile("s_wait_asynccnt 0x0" ::: "memory");
}
__device__ inline void wait_async8() {      // <=8 outstanding: previous chunk landed
  asm volatile("s_wait_asynccnt 0x8" ::: "memory");
}

// ---------- prep kernels ----------
__global__ __launch_bounds__(256) void prep_c_kernel(
    const float* __restrict__ F, const int* __restrict__ t,
    bf16_t* __restrict__ c) {
  int idx = blockIdx.x * 256 + threadIdx.x;
  if (idx >= kM * kD) return;
  int n = (idx >> 7) & (kN - 1);
  int b = idx >> 17;
  float tb = (float)t[b];
  int i = (n < 512) ? n : (n - 512);
  float f = expf(-(float)i * (6.90775527898213705f / 511.0f)); // ln(1000)/(half-1)
  float ang = tb * f;
  float e = (n < 512) ? sinf(ang) : cosf(ang);
  c[idx] = f2bf(F[idx] + e);
}

__global__ __launch_bounds__(256) void cvt_kernel(const float* __restrict__ s,
                                                  bf16_t* __restrict__ d, int n) {
  int i = blockIdx.x * 256 + threadIdx.x;
  if (i < n) d[i] = f2bf(s[i]);
}

// dst[c*R + r] = src[r*C + c]
__global__ __launch_bounds__(256) void tconv_kernel(const float* __restrict__ src,
                                                    int R, int C,
                                                    bf16_t* __restrict__ dst) {
  int idx = blockIdx.x * 256 + threadIdx.x;
  if (idx >= R * C) return;
  int r = idx / C, c = idx - r * C;
  dst[(size_t)c * R + r] = f2bf(src[idx]);
}

// ---------- QKV: [16384x128] @ [128x1024] x3, 32x64 tile per wave ----------
__global__ __launch_bounds__(256) void qkv_kernel(
    const bf16_t* __restrict__ c, const bf16_t* __restrict__ WqT,
    const bf16_t* __restrict__ WkT, const bf16_t* __restrict__ WvT,
    const float* __restrict__ bq, const float* __restrict__ bk,
    const float* __restrict__ bv, bf16_t* __restrict__ qo,
    bf16_t* __restrict__ ko, bf16_t* __restrict__ vTo) {
  const int lane = threadIdx.x & 31;
  const int gw = blockIdx.x * 8 + (threadIdx.x >> 5);
  const int which = gw >> 13;          // 8192 tiles per projection
  const int rem = gw & 8191;
  const int tm = rem >> 4, tn = rem & 15;
  const bf16_t* Bt = (which == 0) ? WqT : (which == 1) ? WkT : WvT;
  const float* bias = (which == 0) ? bq : (which == 1) ? bk : bv;

  v8f acc[8];
#pragma unroll
  for (int i = 0; i < 8; ++i) acc[i] = v8f_zero();
  gemm_2x4(c + (size_t)tm * 32 * kD, kD, Bt + (size_t)tn * 64 * kD, kD,
           kD / 32, lane, acc);

  const int hi = lane >> 4, cl = lane & 15;
#pragma unroll
  for (int ai = 0; ai < 2; ++ai)
#pragma unroll
    for (int ci = 0; ci < 4; ++ci) {
      const int o = tn * 64 + ci * 16 + cl;
      const float bb = bias[o];
      const int h = o >> 7, d = o & 127;
#pragma unroll
      for (int r = 0; r < 8; ++r) {
        int gr = tm * 32 + ai * 16 + 8 * hi + r;
        int b = gr >> 10, n = gr & (kN - 1);
        bf16_t val = f2bf(acc[ai * 4 + ci][r] + bb);
        if (which == 2)
          vTo[(((size_t)b * kH + h) * kD + d) * kN + n] = val;  // V transposed
        else {
          bf16_t* dst = (which == 0) ? qo : ko;
          dst[(((size_t)b * kH + h) * kN + n) * kD + d] = val;
        }
      }
    }
}

// ---------- flash attention: 16 q-rows/wave, 64-key chunks,
//            double-buffered async-to-LDS staging ----------
__global__ __launch_bounds__(256) void flash_attn_kernel(
    const bf16_t* __restrict__ q, const bf16_t* __restrict__ k,
    const bf16_t* __restrict__ vT, bf16_t* __restrict__ ctx) {
  __shared__ __align__(16) bf16_t kl[2 * 64 * 128];   // K tiles [m][d]  2x16KB
  __shared__ __align__(16) bf16_t vl[2 * 128 * 64];   // V tiles [d][m]  2x16KB
  __shared__ __align__(16) bf16_t pl[8 * 16 * 64];    // per-wave P      16KB

  const int tid = threadIdx.x, lane = tid & 31, w = tid >> 5;
  const int bh = blockIdx.x >> 3, rb = blockIdx.x & 7;
  const int n0 = rb * 128 + w * 16;
  const float sc = 0.0883883476483184f;               // 1/sqrt(128)

  v16bf qf[4];
  const size_t qbase = ((size_t)bh * kN + n0) * kD;
#pragma unroll
  for (int s = 0; s < 4; ++s) qf[s] = load_frag_a(q + qbase + s * 32, kD, lane);

  v8f acc[8];
#pragma unroll
  for (int j = 0; j < 8; ++j) acc[j] = v8f_zero();
  v8f mrow, lrow;
#pragma unroll
  for (int r = 0; r < 8; ++r) { mrow[r] = -3.0e38f; lrow[r] = 0.f; }
  bf16_t* plw = pl + w * 1024;

  const unsigned klo = lds_byte_addr(kl);
  const unsigned vlo = lds_byte_addr(vl);
  const int dv = tid >> 1, hf = tid & 1;

  // stage(buf, m0): 8 async b128 per thread (4 for K tile, 4 for V tile)
  auto stage = [&](int buf, int m0) {
    const bf16_t* ks = k + ((size_t)bh * kN + m0) * kD + tid * 32;
    unsigned kd = klo + (unsigned)buf * 16384u + tid * 64;
    async_copy16(kd,      ks);
    async_copy16(kd + 16, ks + 8);
    async_copy16(kd + 32, ks + 16);
    async_copy16(kd + 48, ks + 24);
    const bf16_t* vs = vT + ((size_t)bh * kD + dv) * kN + m0 + hf * 32;
    unsigned vd = vlo + (unsigned)buf * 16384u + (unsigned)(dv * 128 + hf * 64);
    async_copy16(vd,      vs);
    async_copy16(vd + 16, vs + 8);
    async_copy16(vd + 32, vs + 16);
    async_copy16(vd + 48, vs + 24);
  };

  stage(0, 0);                                        // prologue: chunk 0
  for (int cch = 0; cch < kN / 64; ++cch) {
    __syncthreads();                                  // alternate buffer reader-free
    if (cch + 1 < kN / 64) {
      stage((cch + 1) & 1, (cch + 1) * 64);           // prefetch next chunk
      wait_async8();                                  // chunk cch's 8 have landed
    } else {
      wait_async0();
    }
    __syncthreads();                                  // staged data visible to all

    const bf16_t* klc = kl + (cch & 1) * (64 * 128);
    const bf16_t* vlc = vl + (cch & 1) * (128 * 64);

    // S = q . k^T : 16 rows x 64 keys (4 col tiles, K = 128)
    v8f S[4];
#pragma unroll
    for (int cti = 0; cti < 4; ++cti) S[cti] = v8f_zero();
#pragma unroll
    for (int s = 0; s < 4; ++s)
#pragma unroll
      for (int cti = 0; cti < 4; ++cti) {
        v16bf b = load_frag_b(klc + cti * 16 * kD + s * 32, kD, lane);
        S[cti] = wmma_bf16(qf[s], b, S[cti]);
      }

    // online softmax (C-layout: lane = col, element r = row 8*(lane>>4)+r)
    v8f mnew;
#pragma unroll
    for (int r = 0; r < 8; ++r) {
      float a0 = S[0][r] * sc, a1 = S[1][r] * sc;
      float a2 = S[2][r] * sc, a3 = S[3][r] * sc;
      S[0][r] = a0; S[1][r] = a1; S[2][r] = a2; S[3][r] = a3;
      float mx = fmaxf(fmaxf(a0, a1), fmaxf(a2, a3));
      mx = fmaxf(mx, __shfl_xor(mx, 1, 32));
      mx = fmaxf(mx, __shfl_xor(mx, 2, 32));
      mx = fmaxf(mx, __shfl_xor(mx, 4, 32));
      mx = fmaxf(mx, __shfl_xor(mx, 8, 32));
      mnew[r] = fmaxf(mrow[r], mx);
    }
#pragma unroll
    for (int r = 0; r < 8; ++r) {
      float e0 = __expf(S[0][r] - mnew[r]);
      float e1 = __expf(S[1][r] - mnew[r]);
      float e2 = __expf(S[2][r] - mnew[r]);
      float e3 = __expf(S[3][r] - mnew[r]);
      S[0][r] = e0; S[1][r] = e1; S[2][r] = e2; S[3][r] = e3;
      float ps = (e0 + e1) + (e2 + e3);
      ps += __shfl_xor(ps, 1, 32);
      ps += __shfl_xor(ps, 2, 32);
      ps += __shfl_xor(ps, 4, 32);
      ps += __shfl_xor(ps, 8, 32);
      float corr = __expf(mrow[r] - mnew[r]);
      lrow[r] = lrow[r] * corr + ps;
      mrow[r] = mnew[r];
#pragma unroll
      for (int j = 0; j < 8; ++j) acc[j][r] *= corr;
    }

    // transpose P (16x64) through LDS, then P x V(64x128): 16 WMMAs
    {
      const int hi = lane >> 4, cl = lane & 15;
#pragma unroll
      for (int r = 0; r < 8; ++r)
#pragma unroll
        for (int cti = 0; cti < 4; ++cti)
          plw[(8 * hi + r) * 64 + cti * 16 + cl] = f2bf(S[cti][r]);
    }
    v16bf pa0 = load_frag_a(plw, 64, lane);
    v16bf pa1 = load_frag_a(plw + 32, 64, lane);
#pragma unroll
    for (int j = 0; j < 8; ++j) {
      v16bf bv0 = load_frag_b(vlc + j * 16 * 64, 64, lane);
      v16bf bv1 = load_frag_b(vlc + j * 16 * 64 + 32, 64, lane);
      acc[j] = wmma_bf16(pa0, bv0, acc[j]);
      acc[j] = wmma_bf16(pa1, bv1, acc[j]);
    }
  }

  // epilogue: ctx[b, n, h*128+d] = acc / l
  const int b = bh >> 3, h = bh & 7;
  const int hi = lane >> 4, cl = lane & 15;
#pragma unroll
  for (int r = 0; r < 8; ++r) {
    float inv = 1.f / lrow[r];
    int n = n0 + 8 * hi + r;
    size_t base = ((size_t)b * kN + n) * kDH + h * kD + cl;
#pragma unroll
    for (int j = 0; j < 8; ++j) ctx[base + j * 16] = f2bf(acc[j][r] * inv);
  }
}

// ---------- h = ctx @ Wo + bo : [16384x1024] @ [1024x128], 32x64 tiles ----------
__global__ __launch_bounds__(256) void outproj_kernel(
    const bf16_t* __restrict__ ctx, const bf16_t* __restrict__ WoT,
    const float* __restrict__ bo, bf16_t* __restrict__ h) {
  const int lane = threadIdx.x & 31;
  const int gw = blockIdx.x * 8 + (threadIdx.x >> 5);
  const int tm = gw >> 1, tn = gw & 1;
  v8f acc[8];
#pragma unroll
  for (int i = 0; i < 8; ++i) acc[i] = v8f_zero();
  gemm_2x4(ctx + (size_t)tm * 32 * kDH, kDH, WoT + (size_t)tn * 64 * kDH, kDH,
           kDH / 32, lane, acc);
  const int hi = lane >> 4, cl = lane & 15;
#pragma unroll
  for (int ai = 0; ai < 2; ++ai)
#pragma unroll
    for (int ci = 0; ci < 4; ++ci) {
      const int o = tn * 64 + ci * 16 + cl;
      const float bb = bo[o];
#pragma unroll
      for (int r = 0; r < 8; ++r)
        h[(size_t)(tm * 32 + ai * 16 + 8 * hi + r) * kD + o] =
            f2bf(acc[ai * 4 + ci][r] + bb);
    }
}

// ---------- g = h @ Wg, stored transposed per batch [b][d][n] ----------
__global__ __launch_bounds__(256) void gcn_g_kernel(
    const bf16_t* __restrict__ h, const bf16_t* __restrict__ WgT,
    bf16_t* __restrict__ gT) {
  const int lane = threadIdx.x & 31;
  const int gw = blockIdx.x * 8 + (threadIdx.x >> 5);
  const int tm = gw >> 1, tn = gw & 1;
  v8f acc[8];
#pragma unroll
  for (int i = 0; i < 8; ++i) acc[i] = v8f_zero();
  gemm_2x4(h + (size_t)tm * 32 * kD, kD, WgT + (size_t)tn * 64 * kD, kD,
           kD / 32, lane, acc);
  const int hi = lane >> 4, cl = lane & 15;
#pragma unroll
  for (int ai = 0; ai < 2; ++ai)
#pragma unroll
    for (int ci = 0; ci < 4; ++ci) {
      const int d = tn * 64 + ci * 16 + cl;
#pragma unroll
      for (int r = 0; r < 8; ++r) {
        int gr = tm * 32 + ai * 16 + 8 * hi + r;
        int b = gr >> 10, n = gr & (kN - 1);
        gT[((size_t)b * kD + d) * kN + n] = f2bf(acc[ai * 4 + ci][r]);
      }
    }
}

// ---------- feat = relu(A_t @ g) per batch, 32x64 tiles ----------
__global__ __launch_bounds__(256) void gcn_feat_kernel(
    const bf16_t* __restrict__ Abf, const bf16_t* __restrict__ gT,
    float* __restrict__ feat) {
  const int lane = threadIdx.x & 31;
  const int gw = blockIdx.x * 8 + (threadIdx.x >> 5);
  const int b = gw >> 6, rem = gw & 63;
  const int tm = rem >> 1, tn = rem & 1;
  v8f acc[8];
#pragma unroll
  for (int i = 0; i < 8; ++i) acc[i] = v8f_zero();
  gemm_2x4(Abf + (size_t)b * kN * kN + (size_t)tm * 32 * kN, kN,
           gT + (size_t)b * kD * kN + (size_t)tn * 64 * kN, kN, kN / 32, lane, acc);
  const int hi = lane >> 4, cl = lane & 15;
#pragma unroll
  for (int ai = 0; ai < 2; ++ai)
#pragma unroll
    for (int ci = 0; ci < 4; ++ci) {
      const int d = tn * 64 + ci * 16 + cl;
#pragma unroll
      for (int r = 0; r < 8; ++r) {
        int n = tm * 32 + ai * 16 + 8 * hi + r;
        feat[((size_t)b * kN + n) * kD + d] = fmaxf(acc[ai * 4 + ci][r], 0.f);
      }
    }
}

// ---------- row normalize (mean, unbiased std) ----------
__global__ __launch_bounds__(256) void rownorm_kernel(
    const float* __restrict__ feat, bf16_t* __restrict__ nrm) {
  const int row = (blockIdx.x * 256 + threadIdx.x) >> 5;
  const int lane = threadIdx.x & 31;
  if (row >= kM) return;
  const float* p = feat + (size_t)row * kD;
  float x[4], s = 0.f, s2 = 0.f;
#pragma unroll
  for (int i = 0; i < 4; ++i) {
    x[i] = p[lane + 32 * i];
    s += x[i];
    s2 += x[i] * x[i];
  }
#pragma unroll
  for (int m = 1; m < 32; m <<= 1) {
    s += __shfl_xor(s, m, 32);
    s2 += __shfl_xor(s2, m, 32);
  }
  float mu = s * (1.f / 128.f);
  float var = (s2 - 128.f * mu * mu) * (1.f / 127.f);
  var = var < 0.f ? 0.f : var;
  float inv = 1.f / (sqrtf(var) + 1e-8f);
#pragma unroll
  for (int i = 0; i < 4; ++i)
    nrm[(size_t)row * kD + lane + 32 * i] = f2bf((x[i] - mu) * inv);
}

// ---------- out = norm @ norm^T / D per batch, 32x64 tiles ----------
__global__ __launch_bounds__(256) void corr_kernel(
    const bf16_t* __restrict__ nrm, float* __restrict__ out) {
  const int lane = threadIdx.x & 31;
  const int gw = blockIdx.x * 8 + (threadIdx.x >> 5);
  const int b = gw >> 9, rem = gw & 511;
  const int tm = rem >> 4, tn = rem & 15;
  const bf16_t* nb = nrm + (size_t)b * kN * kD;
  v8f acc[8];
#pragma unroll
  for (int i = 0; i < 8; ++i) acc[i] = v8f_zero();
  gemm_2x4(nb + (size_t)tm * 32 * kD, kD, nb + (size_t)tn * 64 * kD, kD,
           kD / 32, lane, acc);
  const int hi = lane >> 4, cl = lane & 15;
#pragma unroll
  for (int ai = 0; ai < 2; ++ai)
#pragma unroll
    for (int ci = 0; ci < 4; ++ci)
#pragma unroll
      for (int r = 0; r < 8; ++r)
        out[(size_t)b * kN * kN +
            (size_t)(tm * 32 + ai * 16 + 8 * hi + r) * kN + tn * 64 + ci * 16 + cl] =
            acc[ai * 4 + ci][r] * (1.f / 128.f);
}

// ---------- host launcher ----------
extern "C" void kernel_launch(void* const* d_in, const int* in_sizes, int n_in,
                              void* d_out, int out_size, void* d_ws, size_t ws_size,
                              hipStream_t stream) {
  const float* A_t = (const float*)d_in[0];
  const float* F_c = (const float*)d_in[1];
  const int*   t   = (const int*)d_in[2];
  const float* Wq  = (const float*)d_in[3];
  const float* bq  = (const float*)d_in[4];
  const float* Wk  = (const float*)d_in[5];
  const float* bk  = (const float*)d_in[6];
  const float* Wv  = (const float*)d_in[7];
  const float* bv  = (const float*)d_in[8];
  const float* Wo  = (const float*)d_in[9];
  const float* bo  = (const float*)d_in[10];
  const float* Wg  = (const float*)d_in[11];
  float* out = (float*)d_out;

  char* ws = (char*)d_ws;
  size_t off = 0;
  auto carve = [&](size_t bytes) -> void* {
    off = (off + 255) & ~(size_t)255;
    void* p = ws + off;
    off += bytes;
    return p;
  };
  bf16_t* c_bf = (bf16_t*)carve((size_t)kM * kD * 2);
  bf16_t* Abf  = (bf16_t*)carve((size_t)kBS * kN * kN * 2);
  bf16_t* WqT  = (bf16_t*)carve((size_t)kDH * kD * 2);
  bf16_t* WkT  = (bf16_t*)carve((size_t)kDH * kD * 2);
  bf16_t* WvT  = (bf16_t*)carve((size_t)kDH * kD * 2);
  bf16_t* WoT  = (bf16_t*)carve((size_t)kD * kDH * 2);
  bf16_t* WgT  = (bf16_t*)carve((size_t)kD * kD * 2);
  bf16_t* qb   = (bf16_t*)carve((size_t)kBS * kH * kN * kD * 2);
  bf16_t* kb   = (bf16_t*)carve((size_t)kBS * kH * kN * kD * 2);
  bf16_t* vTb  = (bf16_t*)carve((size_t)kBS * kH * kD * kN * 2);
  bf16_t* ctx  = (bf16_t*)carve((size_t)kM * kDH * 2);
  bf16_t* hb   = (bf16_t*)carve((size_t)kM * kD * 2);
  bf16_t* gT   = (bf16_t*)carve((size_t)kBS * kD * kN * 2);
  float*  feat = (float*)carve((size_t)kM * kD * 4);
  bf16_t* nrm  = (bf16_t*)carve((size_t)kM * kD * 2);

  prep_c_kernel<<<(kM * kD) / 256, 256, 0, stream>>>(F_c, t, c_bf);
  cvt_kernel<<<(kBS * kN * kN) / 256, 256, 0, stream>>>(A_t, Abf, kBS * kN * kN);
  tconv_kernel<<<(kD * kDH) / 256, 256, 0, stream>>>(Wq, kD, kDH, WqT);
  tconv_kernel<<<(kD * kDH) / 256, 256, 0, stream>>>(Wk, kD, kDH, WkT);
  tconv_kernel<<<(kD * kDH) / 256, 256, 0, stream>>>(Wv, kD, kDH, WvT);
  tconv_kernel<<<(kDH * kD) / 256, 256, 0, stream>>>(Wo, kDH, kD, WoT);
  tconv_kernel<<<(kD * kD) / 256, 256, 0, stream>>>(Wg, kD, kD, WgT);

  qkv_kernel<<<(3 * 8192) / 8, 256, 0, stream>>>(c_bf, WqT, WkT, WvT, bq, bk, bv,
                                                 qb, kb, vTb);
  flash_attn_kernel<<<kBS * kH * (kN / 128), 256, 0, stream>>>(qb, kb, vTb, ctx);
  outproj_kernel<<<(kM / 32) * (kD / 64) / 8, 256, 0, stream>>>(ctx, WoT, bo, hb);
  gcn_g_kernel<<<(kM / 32) * (kD / 64) / 8, 256, 0, stream>>>(hb, WgT, gT);
  gcn_feat_kernel<<<(kBS * (kN / 32) * (kD / 64)) / 8, 256, 0, stream>>>(Abf, gT, feat);
  rownorm_kernel<<<kM / 8, 256, 0, stream>>>(feat, nrm);
  corr_kernel<<<(kBS * (kN / 32) * (kN / 64)) / 8, 256, 0, stream>>>(nrm, out);
}
